// LSTM_12790412607667
// MI455X (gfx1250) — matrix operand order repaired and checked by reading, not compiled, not run on hardware
//
#include <hip/hip_runtime.h>

typedef __attribute__((ext_vector_type(16))) __bf16 v16bf;
typedef __attribute__((ext_vector_type(8)))  __bf16 v8bf;
typedef __attribute__((ext_vector_type(8)))  float  v8f;

#define T_STEPS 128
#define HID     512
#define G4      2048
#define BTILE   16
#define NCLS    10

// ---------------------------------------------------------------------------
// Prep: Wh fp32 -> bf16 (L2-resident operand for WMMA) and the 3x2048 input-
// projection LUT  gx_lut[d][g] = emb[d][0]*Wx[g][0] + emb[d][1]*Wx[g][1] + b[g]
// ---------------------------------------------------------------------------
__global__ __launch_bounds__(512)
void lstm_prep(const float* __restrict__ Wh, const float* __restrict__ emb,
               const float* __restrict__ Wx, const float* __restrict__ bias,
               __bf16* __restrict__ whbf, float* __restrict__ lut)
{
    int idx = blockIdx.x * 512 + threadIdx.x;
    if (idx < G4 * HID) whbf[idx] = (__bf16)Wh[idx];
    if (idx < 3 * G4) {
        int d = idx / G4, g = idx % G4;
        lut[idx] = emb[d * 2 + 0] * Wx[g * 2 + 0] +
                   emb[d * 2 + 1] * Wx[g * 2 + 1] + bias[g];
    }
}

__device__ __forceinline__ float fast_sigmoid(float x) {
    return 1.0f / (1.0f + __expf(-x));
}
__device__ __forceinline__ float fast_tanh(float x) {
    // tanh(x) = 2*sigmoid(2x) - 1  -> single v_exp_f32
    return 2.0f / (1.0f + __expf(-2.0f * x)) - 1.0f;
}

// ---------------------------------------------------------------------------
// Core LSTM: 64 WGs x 16 batch rows, 16 waves/WG, wave owns 32 hidden cols
// (x 4 gates = 8 column tiles).  Per step/wave: 128 v_wmma_f32_16x16x32_bf16.
// h double-buffered in LDS (bf16); c in fp32 registers; one barrier/step.
// ---------------------------------------------------------------------------
__global__ __launch_bounds__(512, 1)
void lstm_core(const int* __restrict__ x, const __bf16* __restrict__ whbf,
               const float* __restrict__ lut, const float* __restrict__ Wp,
               const float* __restrict__ bp, float* __restrict__ out)
{
    __shared__ __bf16 hbuf[2][BTILE][HID];   // 32 KB double-buffered h (bf16)
    __shared__ float  parr[BTILE][NCLS];

    const int tid   = threadIdx.x;
    const int lane  = tid & 31;
    const int wv    = tid >> 5;             // wave 0..15
    const int hi    = (lane >> 4) & 1;      // half-wave select
    const int ln    = lane & 15;
    const int rowbase = blockIdx.x * BTILE;
    const int jbase   = wv * 32;            // this wave's hidden-column slice

    // zero the h buffer read at t=0 (buffer 1)
    {
        unsigned* p = (unsigned*)&hbuf[1][0][0];
        for (int i = tid; i < BTILE * HID / 2; i += 512) p[i] = 0u;
    }

    // Per-lane constants: LUT values for all 3 digits at each of the 8 column
    // tiles, and the Wh row pointers (z-column n == Wh row n, K contiguous).
    float G0[8], G1[8], G2[8];
    const __bf16* bptr[8];
#pragma unroll
    for (int ct = 0; ct < 8; ++ct) {
        int col = (ct >> 1) * HID + jbase + (ct & 1) * 16 + ln;
        G0[ct]  = lut[0 * G4 + col];
        G1[ct]  = lut[1 * G4 + col];
        G2[ct]  = lut[2 * G4 + col];
        bptr[ct] = whbf + (size_t)col * HID;
    }

    v8f cst[2];                              // cell state (fp32, C-layout)
#pragma unroll
    for (int jt = 0; jt < 2; ++jt)
#pragma unroll
        for (int r = 0; r < 8; ++r) cst[jt][r] = 0.0f;

    __syncthreads();

    for (int t = 0; t < T_STEPS; ++t) {
        // digits for the 16 batch rows; broadcast per accumulator row M
        int dig = x[(rowbase + ln) * T_STEPS + t];
        int digv[8];
#pragma unroll
        for (int r = 0; r < 8; ++r)
            digv[r] = __builtin_amdgcn_ds_bpermute((r + 8 * hi) * 4, dig);

        // init accumulators with the input projection (LUT select)
        v8f acc[8];
#pragma unroll
        for (int ct = 0; ct < 8; ++ct)
#pragma unroll
            for (int r = 0; r < 8; ++r) {
                int d = digv[r];
                acc[ct][r] = (d == 0) ? G0[ct] : (d == 1) ? G1[ct] : G2[ct];
            }

        // ---- recurrent GEMM: z += h(t-1) @ Wh^T, K=512 in 16 slabs of 32 ----
        const int rb = (t & 1) ^ 1;                       // read buffer
        const __bf16* arow = &hbuf[rb][ln][0];            // A row M = lane&15
        for (int kb = 0; kb < 16; ++kb) {
            // A fragment (16-bit A layout: lo K = tsel*8.., hi K = +16)
            v16bf av;
            {
                const __bf16* ap = arow + kb * 32 + hi * 8;
                ((v8bf*)&av)[0] = *(const v8bf*)(ap);       // ds_load_b128
                ((v8bf*)&av)[1] = *(const v8bf*)(ap + 16);  // ds_load_b128
            }
#pragma unroll
            for (int ct = 0; ct < 8; ++ct) {
                // B fragment: lane = column n, lo K = hi*8.., hi K = +16
                v16bf bv;
                const __bf16* bpp = bptr[ct] + kb * 32 + hi * 8;
                ((v8bf*)&bv)[0] = *(const v8bf*)(bpp);      // global_load_b128
                ((v8bf*)&bv)[1] = *(const v8bf*)(bpp + 16); // global_load_b128
                acc[ct] = __builtin_amdgcn_wmma_f32_16x16x32_bf16(
                    false, av, false, bv, (short)0, acc[ct], false, false);
            }
        }

        // ---- gate activations, state update, h write-back (bf16) ----
        const int wb = t & 1;
#pragma unroll
        for (int jt = 0; jt < 2; ++jt) {
            v8f gg = acc[0 + jt], ii = acc[2 + jt], ff = acc[4 + jt], oo = acc[6 + jt];
#pragma unroll
            for (int r = 0; r < 8; ++r) {
                float gv = fast_tanh(gg[r]);
                float iv = fast_sigmoid(ii[r]);
                float fv = fast_sigmoid(ff[r]);
                float ov = fast_sigmoid(oo[r]);
                float cv = gv * iv + cst[jt][r] * fv;
                cst[jt][r] = cv;
                float hv = fast_tanh(cv) * ov;
                hbuf[wb][r + 8 * hi][jbase + jt * 16 + ln] = (__bf16)hv;
            }
        }
        __syncthreads();   // double-buffered h: one barrier per step
    }

    // ---- final projection p = h @ Wp + bp, then log_softmax ----
    // final h is in hbuf[(T_STEPS-1)&1] == hbuf[1]
    if (tid < BTILE * NCLS) {
        int row = tid / NCLS, cls = tid % NCLS;
        float s = bp[cls];
        for (int k = 0; k < HID; ++k)
            s += (float)hbuf[1][row][k] * Wp[k * NCLS + cls];
        parr[row][cls] = s;
    }
    __syncthreads();
    if (tid < BTILE) {
        float m = -1e30f;
#pragma unroll
        for (int c2 = 0; c2 < NCLS; ++c2) m = fmaxf(m, parr[tid][c2]);
        float s = 0.0f;
#pragma unroll
        for (int c2 = 0; c2 < NCLS; ++c2) s += __expf(parr[tid][c2] - m);
        float ls = __logf(s);
#pragma unroll
        for (int c2 = 0; c2 < NCLS; ++c2)
            out[(rowbase + tid) * NCLS + c2] = parr[tid][c2] - m - ls;
    }
}

// ---------------------------------------------------------------------------
extern "C" void kernel_launch(void* const* d_in, const int* in_sizes, int n_in,
                              void* d_out, int out_size, void* d_ws, size_t ws_size,
                              hipStream_t stream) {
    const int*   x   = (const int*)d_in[0];
    const float* emb = (const float*)d_in[1];
    const float* Wx  = (const float*)d_in[2];
    const float* Wh  = (const float*)d_in[3];
    const float* b   = (const float*)d_in[4];
    const float* Wp  = (const float*)d_in[5];
    const float* bp  = (const float*)d_in[6];
    float* out = (float*)d_out;

    __bf16* whbf = (__bf16*)d_ws;                                  // 2 MB
    float*  lut  = (float*)((char*)d_ws + (size_t)G4 * HID * 2);   // 24 KB

    lstm_prep<<<(G4 * HID + 511) / 512, 512, 0, stream>>>(Wh, emb, Wx, b, whbf, lut);
    lstm_core<<<1024 / BTILE, 512, 0, stream>>>(x, whbf, lut, Wp, bp, out);
}
